// SignalAttn_86225763435342
// MI455X (gfx1250) — matrix operand order repaired and checked by reading, not compiled, not run on hardware
//
#include <hip/hip_runtime.h>

typedef __attribute__((ext_vector_type(16))) __bf16 bf16x16;
typedef __attribute__((ext_vector_type(8)))  __bf16 bf16x8;
typedef __attribute__((ext_vector_type(8)))  float  floatx8;
typedef __attribute__((ext_vector_type(4)))  float  floatx4;
typedef __attribute__((ext_vector_type(4)))  unsigned uintx4;

#define DEV static __device__ __forceinline__

DEV __bf16 f2bf(float f) {
  unsigned u = __builtin_bit_cast(unsigned, f);
  u += 0x7FFFu + ((u >> 16) & 1u);              // round-to-nearest-even
  unsigned short h = (unsigned short)(u >> 16);
  return __builtin_bit_cast(__bf16, h);
}

DEV unsigned pack2bf(float a, float b) {        // {lo=a, hi=b} as packed bf16x2
  unsigned ua = __builtin_bit_cast(unsigned, a);
  ua += 0x7FFFu + ((ua >> 16) & 1u);
  unsigned ub = __builtin_bit_cast(unsigned, b);
  ub += 0x7FFFu + ((ub >> 16) & 1u);
  return (ua >> 16) | (ub & 0xFFFF0000u);
}

DEV bf16x16 cat8(bf16x8 lo, bf16x8 hi) {
  return __builtin_shufflevector(lo, hi, 0,1,2,3,4,5,6,7,8,9,10,11,12,13,14,15);
}

// Async DMA: global -> LDS, 16 bytes, tracked by ASYNCcnt (CDNA5).
DEV void async_ld_b128(const void* gsrc, void* ldst) {
  unsigned loff = (unsigned)(uintptr_t)ldst;    // flat LDS addr truncates to LDS offset
  asm volatile("global_load_async_to_lds_b128 %0, %1, off"
               :: "v"(loff), "v"(gsrc) : "memory");
}
DEV void wait_async0() { asm volatile("s_wait_asynccnt 0x0" ::: "memory"); }

// 16x32 bf16 A-fragment (ISA 7.12.2): lane m=l&15, K = (l>>4)*8 + {0..7, 16..23}
DEV bf16x16 load_a_frag(const __bf16* base, int ld, int lane) {
  const int m  = lane & 15;
  const int kb = (lane >> 4) * 8;
  const __bf16* p = base + m * ld + kb;
  bf16x8 lo = *(const bf16x8*)(p);
  bf16x8 hi = *(const bf16x8*)(p + 16);
  return cat8(lo, hi);
}

// 32x16 bf16 B-fragment from K-transposed storage bt[n][k]:
// lane n=l&15, K = (l>>4)*8 + {0..7, 16..23}
DEV bf16x16 load_bt_frag(const __bf16* bt, int ld, int lane) {
  const int n  = lane & 15;
  const int kb = (lane >> 4) * 8;
  const __bf16* p = bt + n * ld + kb;
  bf16x8 lo = *(const bf16x8*)(p);
  bf16x8 hi = *(const bf16x8*)(p + 16);
  return cat8(lo, hi);
}

// ---------------------------------------------------------------------------
// C[M,N] = A[M,K] x B[K,N]; A is f32 or bf16, B is f32 weights (cvt to bf16),
// output bf16 (to workspace) or f32 (+bias) to d_out. 128x128 tile / block,
// 8 waves in 4x2, each wave 32x64 = 2x4 WMMA subtiles, K-step 32.
// ---------------------------------------------------------------------------
template<bool A_BF16, bool OUT_BF16, bool ADD_BIAS>
__global__ __launch_bounds__(256)
void gemm_bf16_wmma(const void* __restrict__ Ap, const float* __restrict__ B,
                    void* __restrict__ Cp, const float* __restrict__ bias,
                    int M, int N, int K)
{
  __shared__ __bf16 sA[128 * 40];    // A tile, row-major [m][k], stride 40
  __shared__ __bf16 sBt[128 * 40];   // B tile, K-transposed [n][k], stride 40

  const int tid  = threadIdx.x;
  const int lane = tid & 31;
  const int wave = tid >> 5;
  const int wr = wave >> 1, wc = wave & 1;
  const int gm0 = blockIdx.y * 128;
  const int gn0 = blockIdx.x * 128;

  floatx8 acc[2][4];
  const floatx8 zv = {0,0,0,0,0,0,0,0};
#pragma unroll
  for (int i = 0; i < 2; ++i)
#pragma unroll
    for (int j = 0; j < 4; ++j) acc[i][j] = zv;

  const int arow = tid >> 1, acol = (tid & 1) * 16;   // 128 rows x 32 cols
  const int brow = tid >> 3, bcol = (tid & 7) * 16;   // 32 rows x 128 cols

  for (int kt = 0; kt < K; kt += 32) {
    // ---- stage A tile ----
    if constexpr (A_BF16) {
      const __bf16* src = (const __bf16*)Ap + (size_t)(gm0 + arow) * K + kt + acol;
      async_ld_b128(src,     &sA[arow * 40 + acol]);        // DMA, no VGPR round-trip
      async_ld_b128(src + 8, &sA[arow * 40 + acol + 8]);
    } else {
      const float* src = (const float*)Ap + (size_t)(gm0 + arow) * K + kt + acol;
      floatx4 f0 = *(const floatx4*)(src);
      floatx4 f1 = *(const floatx4*)(src + 4);
      floatx4 f2 = *(const floatx4*)(src + 8);
      floatx4 f3 = *(const floatx4*)(src + 12);
      uintx4 p0 = { pack2bf(f0[0],f0[1]), pack2bf(f0[2],f0[3]),
                    pack2bf(f1[0],f1[1]), pack2bf(f1[2],f1[3]) };
      uintx4 p1 = { pack2bf(f2[0],f2[1]), pack2bf(f2[2],f2[3]),
                    pack2bf(f3[0],f3[1]), pack2bf(f3[2],f3[3]) };
      *(uintx4*)&sA[arow * 40 + acol]     = p0;
      *(uintx4*)&sA[arow * 40 + acol + 8] = p1;
    }
    // ---- stage B tile (transposed into [n][k]) ----
    {
      const float* src = B + (size_t)(kt + brow) * N + gn0 + bcol;
      floatx4 f0 = *(const floatx4*)(src);
      floatx4 f1 = *(const floatx4*)(src + 4);
      floatx4 f2 = *(const floatx4*)(src + 8);
      floatx4 f3 = *(const floatx4*)(src + 12);
#pragma unroll
      for (int i = 0; i < 4; ++i) {
        sBt[(bcol +      i) * 40 + brow] = f2bf(f0[i]);
        sBt[(bcol +  4 + i) * 40 + brow] = f2bf(f1[i]);
        sBt[(bcol +  8 + i) * 40 + brow] = f2bf(f2[i]);
        sBt[(bcol + 12 + i) * 40 + brow] = f2bf(f3[i]);
      }
    }
    if (kt + 32 < K)  // hint the next weight tile into cache (global_prefetch_b8)
      __builtin_prefetch(B + (size_t)(kt + 32 + brow) * N + gn0 + bcol, 0, 0);
    if constexpr (A_BF16) wait_async0();
    __syncthreads();

    bf16x16 afrag[2], bfrag[4];
#pragma unroll
    for (int ms = 0; ms < 2; ++ms)
      afrag[ms] = load_a_frag(&sA[(wr * 32 + ms * 16) * 40], 40, lane);
#pragma unroll
    for (int ns = 0; ns < 4; ++ns)
      bfrag[ns] = load_bt_frag(&sBt[(wc * 64 + ns * 16) * 40], 40, lane);
#pragma unroll
    for (int ms = 0; ms < 2; ++ms)
#pragma unroll
      for (int ns = 0; ns < 4; ++ns)
        acc[ms][ns] = __builtin_amdgcn_wmma_f32_16x16x32_bf16(
            false, afrag[ms], false, bfrag[ns], (short)0, acc[ms][ns], false, false);
    __syncthreads();
  }

  // ---- epilogue: C layout = col l&15, row r + 8*(l>>4) ----
  const int chalf = (lane >> 4) & 1;
  const int nn = lane & 15;
#pragma unroll
  for (int ms = 0; ms < 2; ++ms)
#pragma unroll
    for (int ns = 0; ns < 4; ++ns) {
      const int col = gn0 + wc * 64 + ns * 16 + nn;
#pragma unroll
      for (int r = 0; r < 8; ++r) {
        const int row = gm0 + wr * 32 + ms * 16 + r + 8 * chalf;
        const float v = acc[ms][ns][r];
        if constexpr (OUT_BF16) {
          ((__bf16*)Cp)[(size_t)row * N + col] = f2bf(v);
        } else {
          float b = 0.f;
          if constexpr (ADD_BIAS) b = bias[col];
          ((float*)Cp)[(size_t)row * N + col] = v + b;
        }
      }
    }
}

// ---------------------------------------------------------------------------
// Fused thresholded attention. One workgroup = one (b, h, 128-query block).
// 8 waves, each owns 16 query rows. Pass 1: online softmax stats (m, Z).
// Pass 2: w = max(exp(s-m)/Z, t) - t, O = (w V) / sum(w). All matmuls WMMA.
// K (and V) tiles are DOUBLE-BUFFERED: async DMA streams tile t+1 into LDS
// while tile t is consumed by WMMA. One barrier per iteration.
// ---------------------------------------------------------------------------
__global__ __launch_bounds__(256)
void attn_wmma(const __bf16* __restrict__ Q, const __bf16* __restrict__ Kb,
               const __bf16* __restrict__ Vb, __bf16* __restrict__ O)
{
  constexpr float SCALE = 0.125f;    // 64^-0.5
  constexpr float T = 1e-4f;
  constexpr int SEQ = 2048, INNER = 512, HD = 64;
  constexpr int NT = SEQ / 64;       // 32 key tiles

  __shared__ __bf16 sK[2][64 * 72];   // key tile [j][d] (== B layout for QK^T)
  __shared__ __bf16 sVt[2][64 * 72];  // value tile transposed [d][j]
  __shared__ __bf16 sP[8 * 16 * 72];  // per-wave P staging [m][j]

  const int tid = threadIdx.x;
  const int lane = tid & 31;
  const int wave = tid >> 5;
  const int chalf = (lane >> 4) & 1;
  const int nn = lane & 15;

  const int qb = blockIdx.x & 15;
  const int h  = (blockIdx.x >> 4) & 7;
  const int bb = blockIdx.x >> 7;

  const size_t qrow0 = (size_t)bb * SEQ + qb * 128 + wave * 16;
  const __bf16* qbase = Q  + qrow0 * INNER + h * HD;
  const __bf16* kbase = Kb + (size_t)bb * SEQ * INNER + h * HD;
  const __bf16* vbase = Vb + (size_t)bb * SEQ * INNER + h * HD;

  // Q fragments held in registers for the whole kernel (16x64 per wave)
  bf16x16 qf[2];
#pragma unroll
  for (int ks = 0; ks < 2; ++ks)
    qf[ks] = load_a_frag(qbase + ks * 32, INNER, lane);

  const int srow = tid >> 2;        // 0..63 : tile row
  const int scol = (tid & 3) * 16;  // 0/16/32/48 : col chunk

  float mrow[8], zrow[8];
#pragma unroll
  for (int r = 0; r < 8; ++r) { mrow[r] = -1e30f; zrow[r] = 0.f; }

  const floatx8 zv = {0,0,0,0,0,0,0,0};

  // ---------------- pass 1: softmax statistics ----------------
  {
    const __bf16* src = kbase + (size_t)srow * INNER + scol;
    async_ld_b128(src,     &sK[0][srow * 72 + scol]);
    async_ld_b128(src + 8, &sK[0][srow * 72 + scol + 8]);
    wait_async0();
    __syncthreads();
  }
  for (int t = 0; t < NT; ++t) {
    const int cur = t & 1, nxt = cur ^ 1;
    if (t + 1 < NT) {  // stream next K tile while computing this one
      const __bf16* src = kbase + (size_t)((t + 1) * 64 + srow) * INNER + scol;
      async_ld_b128(src,     &sK[nxt][srow * 72 + scol]);
      async_ld_b128(src + 8, &sK[nxt][srow * 72 + scol + 8]);
    }

    floatx8 s[4];
#pragma unroll
    for (int jt = 0; jt < 4; ++jt) s[jt] = zv;
#pragma unroll
    for (int jt = 0; jt < 4; ++jt)
#pragma unroll
      for (int ks = 0; ks < 2; ++ks) {
        bf16x16 bf = load_bt_frag(&sK[cur][(jt * 16) * 72 + ks * 32], 72, lane);
        s[jt] = __builtin_amdgcn_wmma_f32_16x16x32_bf16(
            false, qf[ks], false, bf, (short)0, s[jt], false, false);
      }
#pragma unroll
    for (int r = 0; r < 8; ++r) {
      float mx = -1e30f;
#pragma unroll
      for (int jt = 0; jt < 4; ++jt) mx = fmaxf(mx, s[jt][r] * SCALE);
#pragma unroll
      for (int off = 8; off >= 1; off >>= 1) mx = fmaxf(mx, __shfl_xor(mx, off, 16));
      const float mnew = fmaxf(mrow[r], mx);
      float ss = 0.f;
#pragma unroll
      for (int jt = 0; jt < 4; ++jt) ss += __expf(s[jt][r] * SCALE - mnew);
#pragma unroll
      for (int off = 8; off >= 1; off >>= 1) ss += __shfl_xor(ss, off, 16);
      zrow[r] = zrow[r] * __expf(mrow[r] - mnew) + ss;
      mrow[r] = mnew;
    }
    wait_async0();
    __syncthreads();
  }

  float zinv[8];
#pragma unroll
  for (int r = 0; r < 8; ++r) zinv[r] = 1.f / zrow[r];

  floatx8 oacc[4];
#pragma unroll
  for (int d = 0; d < 4; ++d) oacc[d] = zv;
  float sw[8];
#pragma unroll
  for (int r = 0; r < 8; ++r) sw[r] = 0.f;

  __bf16* pw = &sP[wave * 16 * 72];

  // ---------------- pass 2: thresholded weights + P*V ----------------
  {   // prologue: stage tile 0 (K via async DMA, V transposed via VGPRs)
    const __bf16* srck = kbase + (size_t)srow * INNER + scol;
    async_ld_b128(srck,     &sK[0][srow * 72 + scol]);
    async_ld_b128(srck + 8, &sK[0][srow * 72 + scol + 8]);
    const __bf16* srcv = vbase + (size_t)srow * INNER + scol;
    bf16x8 va = *(const bf16x8*)srcv;
    bf16x8 vb = *(const bf16x8*)(srcv + 8);
#pragma unroll
    for (int i = 0; i < 8; ++i) {
      sVt[0][(scol + i) * 72 + srow]     = va[i];
      sVt[0][(scol + 8 + i) * 72 + srow] = vb[i];
    }
    wait_async0();
    __syncthreads();
  }
  for (int t = 0; t < NT; ++t) {
    const int cur = t & 1, nxt = cur ^ 1;

    bf16x8 va, vb;
    if (t + 1 < NT) {  // stream next K tile (DMA) and fetch next V rows early
      const __bf16* srck = kbase + (size_t)((t + 1) * 64 + srow) * INNER + scol;
      async_ld_b128(srck,     &sK[nxt][srow * 72 + scol]);
      async_ld_b128(srck + 8, &sK[nxt][srow * 72 + scol + 8]);
      const __bf16* srcv = vbase + (size_t)((t + 1) * 64 + srow) * INNER + scol;
      va = *(const bf16x8*)srcv;
      vb = *(const bf16x8*)(srcv + 8);
    }

    floatx8 s[4];
#pragma unroll
    for (int jt = 0; jt < 4; ++jt) s[jt] = zv;
#pragma unroll
    for (int jt = 0; jt < 4; ++jt)
#pragma unroll
      for (int ks = 0; ks < 2; ++ks) {
        bf16x16 bf = load_bt_frag(&sK[cur][(jt * 16) * 72 + ks * 32], 72, lane);
        s[jt] = __builtin_amdgcn_wmma_f32_16x16x32_bf16(
            false, qf[ks], false, bf, (short)0, s[jt], false, false);
      }

    if (t + 1 < NT) {  // transpose-store next V tile into the other buffer
#pragma unroll
      for (int i = 0; i < 8; ++i) {
        sVt[nxt][(scol + i) * 72 + srow]     = va[i];
        sVt[nxt][(scol + 8 + i) * 72 + srow] = vb[i];
      }
    }

    // w = max(softmax, t) - t ; stage into per-wave LDS region (C->A relayout)
#pragma unroll
    for (int r = 0; r < 8; ++r) {
      float wl = 0.f;
#pragma unroll
      for (int jt = 0; jt < 4; ++jt) {
        const float p = __expf(s[jt][r] * SCALE - mrow[r]) * zinv[r];
        const float w = fmaxf(p, T) - T;
        wl += w;
        pw[(r + 8 * chalf) * 72 + jt * 16 + nn] = f2bf(w);
      }
#pragma unroll
      for (int off = 8; off >= 1; off >>= 1) wl += __shfl_xor(wl, off, 16);
      sw[r] += wl;
    }

    // O += P (16x64) x V (64x64)
#pragma unroll
    for (int d = 0; d < 4; ++d)
#pragma unroll
      for (int ks = 0; ks < 2; ++ks) {
        bf16x16 af = load_a_frag(pw + ks * 32, 72, lane);
        bf16x16 bf = load_bt_frag(&sVt[cur][(d * 16) * 72 + ks * 32], 72, lane);
        oacc[d] = __builtin_amdgcn_wmma_f32_16x16x32_bf16(
            false, af, false, bf, (short)0, oacc[d], false, false);
      }
    wait_async0();
    __syncthreads();
  }

  // write O_inner (bf16) = oacc / sum(w)
#pragma unroll
  for (int d = 0; d < 4; ++d) {
    const int col = h * HD + d * 16 + nn;
#pragma unroll
    for (int r = 0; r < 8; ++r) {
      const size_t row = qrow0 + r + 8 * chalf;
      O[row * INNER + col] = f2bf(oacc[d][r] / sw[r]);
    }
  }
}

// ---------------------------------------------------------------------------
extern "C" void kernel_launch(void* const* d_in, const int* in_sizes, int n_in,
                              void* d_out, int out_size, void* d_ws, size_t ws_size,
                              hipStream_t stream)
{
  const float* fr = (const float*)d_in[0];
  const float* dt = (const float*)d_in[1];
  const float* Wq = (const float*)d_in[2];
  const float* Wk = (const float*)d_in[3];
  const float* Wv = (const float*)d_in[4];
  const float* Wo = (const float*)d_in[5];
  const float* bo = (const float*)d_in[6];

  const int M = 4096, N = 512, K = 512;   // M = b*n rows, N = INNER, K = DIM
  char* ws = (char*)d_ws;
  const size_t qkv_bytes = (size_t)M * N * 2;   // bf16
  __bf16* Qw = (__bf16*)(ws);
  __bf16* Kw = (__bf16*)(ws + 1 * qkv_bytes);
  __bf16* Vw = (__bf16*)(ws + 2 * qkv_bytes);
  __bf16* Ow = (__bf16*)(ws + 3 * qkv_bytes);

  dim3 grid(N / 128, M / 128);
  dim3 blk(256);

  // Q/K/V projections (f32 in -> bf16 out)
  gemm_bf16_wmma<false, true, false><<<grid, blk, 0, stream>>>(fr, Wq, Qw, nullptr, M, N, K);
  gemm_bf16_wmma<false, true, false><<<grid, blk, 0, stream>>>(dt, Wk, Kw, nullptr, M, N, K);
  gemm_bf16_wmma<false, true, false><<<grid, blk, 0, stream>>>(dt, Wv, Vw, nullptr, M, N, K);

  // fused thresholded attention: 2 batches * 8 heads * 16 query blocks = 256
  attn_wmma<<<dim3(256), blk, 0, stream>>>(Qw, Kw, Vw, Ow);

  // output projection with bias (bf16 in -> f32 out)
  gemm_bf16_wmma<true, false, true><<<grid, blk, 0, stream>>>(Ow, Wo, (float*)d_out, bo, M, N, K);

  (void)in_sizes; (void)n_in; (void)out_size; (void)ws_size;
}
